// Net_3813930959289
// MI455X (gfx1250) — matrix operand order repaired and checked by reading, not compiled, not run on hardware
//
#include <hip/hip_runtime.h>
#include <hip/hip_bf16.h>
#include <stdint.h>

typedef __attribute__((ext_vector_type(16))) _Float16 v16h;
typedef __attribute__((ext_vector_type(8)))  _Float16 v8h;
typedef __attribute__((ext_vector_type(2)))  _Float16 h2;
typedef __attribute__((ext_vector_type(8)))  float    v8f;

namespace {
constexpr int RESN = 512;
constexpr int NCH  = 32;    // EMB_DIM
constexpr int HID  = 128;
constexpr int BLK  = 128;   // threads per block = 4 waves
constexpr int FSTR = 40;    // feats LDS row stride (halves), 80B = 5x16B
constexpr int HSTR = 136;   // hidden LDS row stride (halves), 272B = 17x16B
// packed-weights layout (u32 units):
//   layer0: t(0..7) * 256 + lane*8 + v                      -> 2048 u32
//   layer1: 2048  + (t*4+kc)*256 + lane*8 + v               -> 8192 u32
//   layer2: 10240 + (t*4+kc)*256 + lane*8 + v               -> 8192 u32
constexpr int PACK_TOTAL = 18432;
constexpr size_t P16_OFF   = 131072;                      // f16 planes offset in ws
constexpr size_t P16_BYTES = (size_t)3 * RESN * RESN * NCH * 2;  // 50,331,648
constexpr int PIX_TOTAL = 3 * RESN * RESN;                // 786432 pixels
}

struct B32x2 { uint4 lo, hi; };
static __device__ __forceinline__ v16h make_frag(uint4 lo, uint4 hi) {
  B32x2 t{lo, hi};
  return __builtin_bit_cast(v16h, t);
}
static __device__ __forceinline__ h2 pkrtz(float a, float b) {
  auto r = __builtin_amdgcn_cvt_pkrtz(a, b);   // __fp16 vec2
  return __builtin_bit_cast(h2, r);
}
// bias-add + ReLU in packed f16 (inputs canonical: from cvt/add -> no
// canonicalize ops; lowers to v_pk_add_f16 + v_pk_max_num_f16)
static __device__ __forceinline__ h2 bias_relu(h2 v, h2 bias2) {
  h2 s = v + bias2;
  h2 z = {};
  return __builtin_elementwise_max(s, z);
}

// ---------------------------------------------------------------------------
// Pack w0/w1/w2 (f32, [out][in]) into WMMA B-fragment order as f16 pairs.
// B 32x16 (KxN) 16-bit layout: lanes 0-15 -> N=lane, K=0..7,16..23 (2/VGPR);
//                              lanes 16-31 -> N=lane-16, K=8..15,24..31.
// ---------------------------------------------------------------------------
__global__ void pack_weights(const float* __restrict__ w0,
                             const float* __restrict__ w1,
                             const float* __restrict__ w2,
                             uint32_t* __restrict__ ws) {
  int idx = blockIdx.x * 256 + threadIdx.x;
  if (idx >= PACK_TOTAL) return;
  const float* w; int in, t, kc, r;
  if (idx < 2048)       { w = w0; in = NCH; t = idx >> 8;  kc = 0;              r = idx & 255; }
  else if (idx < 10240) { int j = idx - 2048;  w = w1; in = HID; t = j >> 10; kc = (j >> 8) & 3; r = j & 255; }
  else                  { int j = idx - 10240; w = w2; in = HID; t = j >> 10; kc = (j >> 8) & 3; r = j & 255; }
  int lane = r >> 3, v = r & 7;
  int n = 16 * t + (lane & 15);
  int k = kc * 32 + ((lane >> 4) & 1) * 8 + (v >> 2) * 16 + 2 * (v & 3);
  h2 pr;
  pr[0] = (_Float16)w[n * in + k];
  pr[1] = (_Float16)w[n * in + k + 1];
  ws[idx] = __builtin_bit_cast(uint32_t, pr);
}

// ---------------------------------------------------------------------------
// Repack planes [3][32][512][512] f32 -> [3][512][512][32] f16 (channel-last).
// Reads are wave-coalesced per channel; writes are 64B contiguous per lane.
// ---------------------------------------------------------------------------
__global__ void repack_planes(const float* __restrict__ planes,
                              _Float16* __restrict__ out) {
  int idx = blockIdx.x * 256 + threadIdx.x;   // pl*262144 + y*512 + x
  if (idx >= PIX_TOTAL) return;
  int pl = idx >> 18;
  int rem = idx & (RESN * RESN - 1);
  uint32_t buf[NCH / 2];
#pragma unroll
  for (int c = 0; c < NCH; c += 2) {
    h2 pr;
    pr[0] = (_Float16)planes[((size_t)(pl * NCH + c))     * (RESN * RESN) + rem];
    pr[1] = (_Float16)planes[((size_t)(pl * NCH + c + 1)) * (RESN * RESN) + rem];
    buf[c >> 1] = __builtin_bit_cast(uint32_t, pr);
  }
  uint4* o = (uint4*)(out + (size_t)idx * NCH);
#pragma unroll
  for (int i = 0; i < 4; ++i) o[i] = ((const uint4*)buf)[i];
}

// ---------------------------------------------------------------------------
// Main fused kernel: tri-plane bilinear sample -> 3x WMMA layers -> dot out.
// MODE==1: sample from channel-last f16 planes (4x b128 per tap).
// MODE==0: fallback, sample original channel-major f32 planes.
// ---------------------------------------------------------------------------
template <int MODE>
__global__ __launch_bounds__(BLK)
void triplane_mlp(const float* __restrict__ coords,
                  const float* __restrict__ planes,
                  const _Float16* __restrict__ planes16,
                  const uint32_t* __restrict__ wp,
                  const float* __restrict__ b0,
                  const float* __restrict__ b1,
                  const float* __restrict__ b2,
                  const float* __restrict__ w3,
                  const float* __restrict__ b3,
                  float* __restrict__ out, int n) {
  __shared__ _Float16 s_feat[BLK * FSTR];
  __shared__ _Float16 s_h[BLK * HSTR];

  const int tid  = threadIdx.x;
  const int lane = tid & 31;
  const int wave = tid >> 5;
  const int rowbase = wave * 32;
  const int p  = blockIdx.x * BLK + tid;
  const int pr = (p < n) ? p : 0;

  // ---- Phase 1: bilinear tri-plane sampling (1 lane = 1 point) ----
  float f[NCH];
#pragma unroll
  for (int c = 0; c < NCH; ++c) f[c] = 0.f;

  const float cx = coords[3 * pr + 0];
  const float cy = coords[3 * pr + 1];
  const float cz = coords[3 * pr + 2];
  const float px[3] = {cx, cy, cx};
  const float py[3] = {cy, cz, cz};

#pragma unroll
  for (int pl = 0; pl < 3; ++pl) {
    float x = (px[pl] + 1.f) * 0.5f * (RESN - 1);
    float y = (py[pl] + 1.f) * 0.5f * (RESN - 1);
    float x0f = floorf(x), y0f = floorf(y);
    int x0 = (int)x0f, y0 = (int)y0f;
    float wx1 = x - x0f, wy1 = y - y0f;
    float wx0 = 1.f - wx1, wy0 = 1.f - wy1;
    float wt[4] = {wx0 * wy0, wx1 * wy0, wx0 * wy1, wx1 * wy1};
    int xc4[4], yc4[4];
#pragma unroll
    for (int t = 0; t < 4; ++t) {
      int xi = x0 + (t & 1), yi = y0 + (t >> 1);
      bool ok = (xi >= 0) && (xi < RESN) && (yi >= 0) && (yi < RESN);
      xc4[t] = xi < 0 ? 0 : (xi > RESN - 1 ? RESN - 1 : xi);
      yc4[t] = yi < 0 ? 0 : (yi > RESN - 1 ? RESN - 1 : yi);
      wt[t] = ok ? wt[t] : 0.f;
    }
    if constexpr (MODE == 1) {
      const _Float16* pb = planes16 + (size_t)pl * (RESN * RESN) * NCH;
#pragma unroll
      for (int t = 0; t < 4; ++t) {
        const uint4* tp = (const uint4*)(pb + ((size_t)yc4[t] * RESN + xc4[t]) * NCH);
        const float w = wt[t];
#pragma unroll
        for (int i = 0; i < 4; ++i) {
          v8h hv = __builtin_bit_cast(v8h, tp[i]);
#pragma unroll
          for (int j = 0; j < 8; ++j) f[i * 8 + j] += w * (float)hv[j];
        }
      }
    } else {
      const float* bp[4];
#pragma unroll
      for (int t = 0; t < 4; ++t)
        bp[t] = planes + (size_t)pl * NCH * RESN * RESN + (size_t)yc4[t] * RESN + xc4[t];
#pragma unroll
      for (int c = 0; c < NCH; ++c) {
        size_t co = (size_t)c * RESN * RESN;
        f[c] += wt[0] * bp[0][co] + wt[1] * bp[1][co]
              + wt[2] * bp[2][co] + wt[3] * bp[3][co];
      }
    }
  }

  // stage features to LDS as f16 (row-major [point][ch], padded rows)
  {
    uint32_t* fr = (uint32_t*)((char*)s_feat + (size_t)tid * (FSTR * 2));
#pragma unroll
    for (int c = 0; c < NCH; c += 2) {
      h2 hp = pkrtz(f[c], f[c + 1]);
      fr[c >> 1] = __builtin_bit_cast(uint32_t, hp);
    }
  }
  __syncthreads();

  const int laneOff = (lane >= 16) ? 16 : 0;  // byte offset for A-frag K groups
  const int col     = lane & 15;              // N / M-sub index
  const int rowAdd  = (lane >= 16) ? 8 : 0;   // C/D row offset

  // ---- Layer 0: [16x32] x [32x128]; one B load feeds both M-tiles ----
  {
    v16h A[2];
#pragma unroll
    for (int m = 0; m < 2; ++m) {
      const int arow = rowbase + m * 16 + col;
      const char* ap = (const char*)s_feat + (size_t)arow * (FSTR * 2) + laneOff;
      A[m] = make_frag(*(const uint4*)ap, *(const uint4*)(ap + 32));
    }
#pragma unroll
    for (int t = 0; t < 8; ++t) {
      const uint4* bw = (const uint4*)(wp + (size_t)(t * 32 + lane) * 8);
      v16h B = make_frag(bw[0], bw[1]);
      float biasf = b0[t * 16 + col];
      h2 bias2 = pkrtz(biasf, biasf);
      v8f C0 = {}, C1 = {};   // SRC2 = inline constant 0
      C0 = __builtin_amdgcn_wmma_f32_16x16x32_f16(false, A[0], false, B,
                                                  (short)0, C0, false, false);
      C1 = __builtin_amdgcn_wmma_f32_16x16x32_f16(false, A[1], false, B,
                                                  (short)0, C1, false, false);
      const int pt0 = rowbase + rowAdd;
#pragma unroll
      for (int r = 0; r < 8; r += 2) {
        h2 h0 = bias_relu(pkrtz(C0[r], C0[r + 1]), bias2);
        h2 h1 = bias_relu(pkrtz(C1[r], C1[r + 1]), bias2);
        s_h[(size_t)(pt0 + r) * HSTR + t * 16 + col]          = h0[0];
        s_h[(size_t)(pt0 + r + 1) * HSTR + t * 16 + col]      = h0[1];
        s_h[(size_t)(pt0 + 16 + r) * HSTR + t * 16 + col]     = h1[0];
        s_h[(size_t)(pt0 + 16 + r + 1) * HSTR + t * 16 + col] = h1[1];
      }
    }
  }

  // ---- Layers 1 & 2: [16x128] x [128x128]; K chained, B shared across M ----
#pragma unroll
  for (int L = 0; L < 2; ++L) {
    const uint32_t* lw = wp + 2048 + (size_t)L * 8192;
    const float* bb = L ? b2 : b1;
    v16h A[2][4];
#pragma unroll
    for (int m = 0; m < 2; ++m)
#pragma unroll
      for (int kc = 0; kc < 4; ++kc) {
        const int arow = rowbase + m * 16 + col;
        const char* ap = (const char*)s_h + (size_t)arow * (HSTR * 2) + kc * 64 + laneOff;
        A[m][kc] = make_frag(*(const uint4*)ap, *(const uint4*)(ap + 32));
      }
#pragma unroll
    for (int t = 0; t < 8; ++t) {
      float biasf = bb[t * 16 + col];
      h2 bias2 = pkrtz(biasf, biasf);
      v8f C0 = {}, C1 = {};   // SRC2 = inline constant 0
#pragma unroll
      for (int kc = 0; kc < 4; ++kc) {
        const uint4* bw = (const uint4*)(lw + (size_t)((t * 4 + kc) * 32 + lane) * 8);
        v16h B = make_frag(bw[0], bw[1]);
        C0 = __builtin_amdgcn_wmma_f32_16x16x32_f16(false, A[0][kc], false, B,
                                                    (short)0, C0, false, false);
        C1 = __builtin_amdgcn_wmma_f32_16x16x32_f16(false, A[1][kc], false, B,
                                                    (short)0, C1, false, false);
      }
      const int pt0 = rowbase + rowAdd;
#pragma unroll
      for (int r = 0; r < 8; r += 2) {
        h2 h0 = bias_relu(pkrtz(C0[r], C0[r + 1]), bias2);
        h2 h1 = bias_relu(pkrtz(C1[r], C1[r + 1]), bias2);
        s_h[(size_t)(pt0 + r) * HSTR + t * 16 + col]          = h0[0];
        s_h[(size_t)(pt0 + r + 1) * HSTR + t * 16 + col]      = h0[1];
        s_h[(size_t)(pt0 + 16 + r) * HSTR + t * 16 + col]     = h1[0];
        s_h[(size_t)(pt0 + 16 + r + 1) * HSTR + t * 16 + col] = h1[1];
      }
    }
  }

  // ---- Layer 3: 128 -> 1 per-lane dot product ----
  float o = b3[0];
  const uint4* hr = (const uint4*)((const char*)s_h + (size_t)tid * (HSTR * 2));
#pragma unroll
  for (int i = 0; i < 16; ++i) {
    uint4 q = hr[i];
    v8h hv = __builtin_bit_cast(v8h, q);
#pragma unroll
    for (int j = 0; j < 8; ++j)
      o += (float)hv[j] * w3[i * 8 + j];
  }
  if (p < n) out[p] = o;
}

// ---------------------------------------------------------------------------
extern "C" void kernel_launch(void* const* d_in, const int* in_sizes, int n_in,
                              void* d_out, int out_size, void* d_ws, size_t ws_size,
                              hipStream_t stream) {
  const float* coords = (const float*)d_in[0];
  const float* planes = (const float*)d_in[1];
  const float* w0 = (const float*)d_in[2];
  const float* b0 = (const float*)d_in[3];
  const float* w1 = (const float*)d_in[4];
  const float* b1 = (const float*)d_in[5];
  const float* w2 = (const float*)d_in[6];
  const float* b2 = (const float*)d_in[7];
  const float* w3 = (const float*)d_in[8];
  const float* b3 = (const float*)d_in[9];
  uint32_t* wpack = (uint32_t*)d_ws;
  (void)n_in; (void)out_size;

  pack_weights<<<PACK_TOTAL / 256, 256, 0, stream>>>(w0, w1, w2, wpack);

  int n = in_sizes[0] / 3;
  int grid = (n + BLK - 1) / BLK;

  const bool channelLast = ws_size >= (P16_OFF + P16_BYTES);
  if (channelLast) {
    _Float16* p16 = (_Float16*)((char*)d_ws + P16_OFF);
    repack_planes<<<(PIX_TOTAL + 255) / 256, 256, 0, stream>>>(planes, p16);
    triplane_mlp<1><<<grid, BLK, 0, stream>>>(coords, planes, p16, wpack,
                                              b0, b1, b2, w3, b3,
                                              (float*)d_out, n);
  } else {
    triplane_mlp<0><<<grid, BLK, 0, stream>>>(coords, planes, nullptr, wpack,
                                              b0, b1, b2, w3, b3,
                                              (float*)d_out, n);
  }
}
